// MoEFFNLayer_70729521430575
// MI455X (gfx1250) — compile-verified, matrix operand order checked by reading, not run on hardware
//
#include <hip/hip_runtime.h>
#include <hip/hip_bf16.h>

// ---------------- problem constants ----------------
#define NEXP   8
#define TOPK   2
#define HID    1024
#define IDIM   4096
#define TOKENS 8192          // 4 * 2048

// ---------------- tiling ----------------
#define BM     64            // token rows per workgroup
#define NTHREADS 1024        // 32 waves
#define IC     128           // inter-dim chunk per iteration
#define XLD    (HID + 8)     // padded LDS stride (halves): 2064B/row -> bank shift 4
#define ACT_LD (IC + 8)      // 272B/row -> bank shift 4

#define X_BYTES   (BM * XLD * 2)        // 132096
#define ACT_BYTES (BM * ACT_LD * 2)     // 17408
#define SMEM_BYTES (X_BYTES + ACT_BYTES + BM * 4 + BM * 4)   // 150016

#define WELEMS ((size_t)NEXP * IDIM * HID)   // elements per weight tensor (33,554,432)

typedef __attribute__((ext_vector_type(16))) __bf16 v16bf;
typedef __attribute__((ext_vector_type(8)))  __bf16 v8bf;
typedef __attribute__((ext_vector_type(8)))  float  v8f;

#define WMMA_BF16(A, B, C) \
  __builtin_amdgcn_wmma_f32_16x16x32_bf16(false, (A), false, (B), (short)0, (C), false, false)

// A fragment (16x32 bf16) from LDS, row-major with stride ld (halves).
// Lane l: row = l&15, k-halves [hk,hk+8) and [hk+16,hk+24), hk=(l>>4)*8.
__device__ __forceinline__ v16bf load_a_lds(const __bf16* base, int ld, int mrow,
                                            int k0, int lane) {
  const int r  = lane & 15;
  const int hk = (lane >> 4) << 3;
  const __bf16* p = base + (mrow + r) * ld + k0 + hk;
  v8bf lo = *(const v8bf*)(p);
  v8bf hi = *(const v8bf*)(p + 16);
  v16bf a;
#pragma unroll
  for (int i = 0; i < 8; ++i) { a[i] = lo[i]; a[i + 8] = hi[i]; }
  return a;
}

// B fragment (32x16 bf16) from f32 weight rows (on-the-fly convert).
__device__ __forceinline__ v16bf load_b_glb(const float* __restrict__ W, size_t ld,
                                            int nrow, int k0, int lane) {
  const int n  = lane & 15;
  const int hk = (lane >> 4) << 3;
  const float* p = W + (size_t)(nrow + n) * ld + (size_t)(k0 + hk);
  float4 f0 = *(const float4*)(p);
  float4 f1 = *(const float4*)(p + 4);
  float4 f2 = *(const float4*)(p + 16);
  float4 f3 = *(const float4*)(p + 20);
  v16bf b;
  b[0]  = (__bf16)f0.x; b[1]  = (__bf16)f0.y; b[2]  = (__bf16)f0.z; b[3]  = (__bf16)f0.w;
  b[4]  = (__bf16)f1.x; b[5]  = (__bf16)f1.y; b[6]  = (__bf16)f1.z; b[7]  = (__bf16)f1.w;
  b[8]  = (__bf16)f2.x; b[9]  = (__bf16)f2.y; b[10] = (__bf16)f2.z; b[11] = (__bf16)f2.w;
  b[12] = (__bf16)f3.x; b[13] = (__bf16)f3.y; b[14] = (__bf16)f3.z; b[15] = (__bf16)f3.w;
  return b;
}

// B fragment (32x16 bf16) from pre-converted bf16 weight rows: 2x16B loads, no cvt.
__device__ __forceinline__ v16bf load_b_glb(const __bf16* __restrict__ W, size_t ld,
                                            int nrow, int k0, int lane) {
  const int n  = lane & 15;
  const int hk = (lane >> 4) << 3;
  const __bf16* p = W + (size_t)(nrow + n) * ld + (size_t)(k0 + hk);
  v8bf lo = *(const v8bf*)(p);
  v8bf hi = *(const v8bf*)(p + 16);
  v16bf b;
#pragma unroll
  for (int i = 0; i < 8; ++i) { b[i] = lo[i]; b[i + 8] = hi[i]; }
  return b;
}

// ---------------- kernel 0: f32 -> bf16 weight conversion ----------------
__global__ void moe_cvt_bf16_kernel(const float* __restrict__ src,
                                    __bf16* __restrict__ dst, int n8) {
  int i = blockIdx.x * blockDim.x + threadIdx.x;
  const int stride = gridDim.x * blockDim.x;
  for (; i < n8; i += stride) {
    const float4 a = *(const float4*)(src + (size_t)i * 8);
    const float4 b = *(const float4*)(src + (size_t)i * 8 + 4);
    v8bf o;
    o[0] = (__bf16)a.x; o[1] = (__bf16)a.y; o[2] = (__bf16)a.z; o[3] = (__bf16)a.w;
    o[4] = (__bf16)b.x; o[5] = (__bf16)b.y; o[6] = (__bf16)b.z; o[7] = (__bf16)b.w;
    *(v8bf*)(dst + (size_t)i * 8) = o;
  }
}

// ---------------- kernel 1: zero output + counters ----------------
__global__ void moe_zero_kernel(float* __restrict__ out, int n,
                                int* __restrict__ counts, float* __restrict__ sums) {
  int i = blockIdx.x * blockDim.x + threadIdx.x;
  const int stride = gridDim.x * blockDim.x;
  for (; i < n; i += stride) out[i] = 0.0f;
  if (blockIdx.x == 0 && threadIdx.x < NEXP) {
    counts[threadIdx.x] = 0;
    sums[threadIdx.x]   = 0.0f;
  }
}

// ---------------- kernel 2: router (softmax + top-2 + lists) ----------------
__global__ void moe_router_kernel(const float* __restrict__ xf,
                                  const float* __restrict__ gate_w,
                                  float* __restrict__ combine,
                                  int*   __restrict__ counts,
                                  int*   __restrict__ tok_list,
                                  float* __restrict__ sum_probs) {
  __shared__ float sp[NEXP];
  if (threadIdx.x < NEXP) sp[threadIdx.x] = 0.0f;
  __syncthreads();

  const int t = blockIdx.x * blockDim.x + threadIdx.x;
  const float* xr = xf + (size_t)t * HID;

  float lg[NEXP] = {0.f, 0.f, 0.f, 0.f, 0.f, 0.f, 0.f, 0.f};
  for (int k = 0; k < HID; k += 4) {
    float4 xv = *(const float4*)(xr + k);
#pragma unroll
    for (int e = 0; e < NEXP; ++e) {
      float4 gv = *(const float4*)(gate_w + (size_t)e * HID + k);
      lg[e] += xv.x * gv.x + xv.y * gv.y + xv.z * gv.z + xv.w * gv.w;
    }
  }

  float mx = lg[0];
#pragma unroll
  for (int e = 1; e < NEXP; ++e) mx = fmaxf(mx, lg[e]);
  float p[NEXP]; float s = 0.0f;
#pragma unroll
  for (int e = 0; e < NEXP; ++e) { p[e] = __expf(lg[e] - mx); s += p[e]; }
  const float inv = 1.0f / s;
#pragma unroll
  for (int e = 0; e < NEXP; ++e) p[e] *= inv;

  int i0 = 0;
#pragma unroll
  for (int e = 1; e < NEXP; ++e) if (p[e] > p[i0]) i0 = e;
  int i1 = (i0 == 0) ? 1 : 0;
#pragma unroll
  for (int e = 0; e < NEXP; ++e) if (e != i0 && p[e] > p[i1]) i1 = e;

  const float denom = fmaxf(p[i0] + p[i1], 1e-8f);
  const float w0 = p[i0] / denom;
  const float w1 = p[i1] / denom;

#pragma unroll
  for (int e = 0; e < NEXP; ++e)
    combine[(size_t)t * NEXP + e] = (e == i0) ? w0 : ((e == i1) ? w1 : 0.0f);

  int s0 = atomicAdd(&counts[i0], 1); tok_list[(size_t)i0 * TOKENS + s0] = t;
  int s1 = atomicAdd(&counts[i1], 1); tok_list[(size_t)i1 * TOKENS + s1] = t;

#pragma unroll
  for (int e = 0; e < NEXP; ++e) atomicAdd(&sp[e], p[e]);
  __syncthreads();
  if (threadIdx.x < NEXP) unsafeAtomicAdd(&sum_probs[threadIdx.x], sp[threadIdx.x]);
}

// ---------------- kernel 3: aux loss scalar ----------------
__global__ void moe_aux_kernel(const int* __restrict__ counts,
                               const float* __restrict__ sum_probs,
                               float* __restrict__ out_aux) {
  if (threadIdx.x == 0 && blockIdx.x == 0) {
    float imp[NEXP]; float s = 0.0f;
#pragma unroll
    for (int e = 0; e < NEXP; ++e) { imp[e] = sum_probs[e] / (float)TOKENS; s += imp[e]; }
    s = fmaxf(s, 1e-8f);
    float acc = 0.0f;
#pragma unroll
    for (int e = 0; e < NEXP; ++e) {
      float usage = (float)counts[e] / (float)(TOKENS * TOPK);
      acc += usage * (imp[e] / s);
    }
    *out_aux = fminf(acc * (float)NEXP, 1.0f) * 0.01f;
  }
}

// ---------------- kernel 4: fused grouped SwiGLU FFN ----------------
template <typename WT>
__global__ __launch_bounds__(NTHREADS)
void moe_ffn_kernel(const float* __restrict__ xf,
                    const WT* __restrict__ wg,
                    const WT* __restrict__ wu,
                    const WT* __restrict__ wd,
                    const int*   __restrict__ counts,
                    const int*   __restrict__ tok_list,
                    const float* __restrict__ combine,
                    float* __restrict__ out) {
  const int e    = blockIdx.y;
  const int mblk = blockIdx.x;
  const int cnt  = counts[e];
  if (mblk * BM >= cnt) return;

  extern __shared__ char smem[];
  __bf16* x_lds   = (__bf16*)smem;
  __bf16* act_lds = (__bf16*)(smem + X_BYTES);
  int*    tok_lds = (int*)  (smem + X_BYTES + ACT_BYTES);
  float*  w_lds   = (float*)(smem + X_BYTES + ACT_BYTES + BM * 4);

  const int tid  = threadIdx.x;
  const int lane = tid & 31;
  const int wv   = tid >> 5;            // 0..31

  const WT* wg_e = wg + (size_t)e * IDIM * HID;
  const WT* wu_e = wu + (size_t)e * IDIM * HID;
  const WT* wd_e = wd + (size_t)e * HID * IDIM;

  // token ids + combine weights for this row block
  if (tid < BM) {
    const int row = mblk * BM + tid;
    const int t   = (row < cnt) ? tok_list[(size_t)e * TOKENS + row] : 0;
    tok_lds[tid] = t;
    w_lds[tid]   = (row < cnt) ? combine[(size_t)t * NEXP + e] : 0.0f;
  }
  __syncthreads();

  // stage x rows into LDS as bf16 (1024 threads: 64 rows x 16 groups of 64 cols)
  {
    const int row = tid >> 4;
    const int g16 = tid & 15;
    const int t   = tok_lds[row];
    const float* src = xf + (size_t)t * HID + g16 * 64;
    __bf16* dst = x_lds + row * XLD + g16 * 64;
#pragma unroll
    for (int j = 0; j < 16; ++j) {
      float4 v = *(const float4*)(src + j * 4);
      dst[j * 4 + 0] = (__bf16)v.x;
      dst[j * 4 + 1] = (__bf16)v.y;
      dst[j * 4 + 2] = (__bf16)v.z;
      dst[j * 4 + 3] = (__bf16)v.w;
    }
  }
  __syncthreads();

  const int mtile = (wv & 3) * 16;        // token tile of this wave (4 tiles of 16)
  const int ctile = (wv >> 2) * 16;       // stage-1 I-col tile within chunk (8 tiles)
  const int hbase = (wv >> 2) * 128;      // stage-2 H range of this wave

  v8f yacc[8];
#pragma unroll
  for (int ht = 0; ht < 8; ++ht) yacc[ht] = (v8f)(0.0f);

#pragma unroll 1
  for (int ci = 0; ci < IDIM / IC; ++ci) {
    const int i0 = ci * IC;

    // ---- stage 1: G, U tiles over full H, then silu(g)*u -> act_lds ----
    v8f gacc = (v8f)(0.0f);
    v8f uacc = (v8f)(0.0f);
    const int nrow = i0 + ctile;
#pragma unroll 4
    for (int k0 = 0; k0 < HID; k0 += 32) {
      v16bf a  = load_a_lds(x_lds, XLD, mtile, k0, lane);
      v16bf bg = load_b_glb(wg_e, HID, nrow, k0, lane);
      gacc = WMMA_BF16(a, bg, gacc);
      v16bf bu = load_b_glb(wu_e, HID, nrow, k0, lane);
      uacc = WMMA_BF16(a, bu, uacc);
    }
    {
      const int n    = lane & 15;
      const int moff = (lane >> 4) << 3;
#pragma unroll
      for (int v = 0; v < 8; ++v) {
        const float g = gacc[v];
        const float u = uacc[v];
        const float aval = (g / (1.0f + __expf(-g))) * u;   // silu(g) * u
        act_lds[(mtile + moff + v) * ACT_LD + ctile + n] = (__bf16)aval;
      }
    }
    __syncthreads();

    // ---- stage 2: Y += Act_chunk @ wd_chunk^T ----
#pragma unroll
    for (int k0 = 0; k0 < IC; k0 += 32) {
      v16bf a = load_a_lds(act_lds, ACT_LD, mtile, k0, lane);
#pragma unroll
      for (int ht = 0; ht < 8; ++ht) {
        v16bf b = load_b_glb(wd_e, IDIM, hbase + ht * 16, i0 + k0, lane);
        yacc[ht] = WMMA_BF16(a, b, yacc[ht]);
      }
    }
    __syncthreads();
  }

  // ---- scatter: out[t, h] += combine * y ----
  {
    const int n    = lane & 15;
    const int moff = (lane >> 4) << 3;
#pragma unroll
    for (int v = 0; v < 8; ++v) {
      const int   m   = mtile + moff + v;
      const float wgt = w_lds[m];
      if (wgt != 0.0f) {
        const size_t base = (size_t)tok_lds[m] * HID;
#pragma unroll
        for (int ht = 0; ht < 8; ++ht) {
          unsafeAtomicAdd(&out[base + hbase + ht * 16 + n], wgt * yacc[ht][v]);
        }
      }
    }
  }
}

// ---------------- host launcher ----------------
extern "C" void kernel_launch(void* const* d_in, const int* in_sizes, int n_in,
                              void* d_out, int out_size, void* d_ws, size_t ws_size,
                              hipStream_t stream) {
  const float* xf     = (const float*)d_in[0];
  const float* gate_w = (const float*)d_in[1];
  const float* wg     = (const float*)d_in[2];
  const float* wu     = (const float*)d_in[3];
  const float* wd     = (const float*)d_in[4];
  float* out = (float*)d_out;

  // workspace layout
  char*  ws        = (char*)d_ws;
  int*   counts    = (int*)ws;                              // 8 ints
  float* sum_probs = (float*)(ws + 64);                     // 8 floats
  float* combine   = (float*)(ws + 128);                    // T*8 f32 (256 KB)
  int*   tok_list  = (int*)(ws + 128 + (size_t)TOKENS * NEXP * 4); // 8*T ints (256 KB)
  const size_t hdr_end = 128 + (size_t)TOKENS * NEXP * 4 + (size_t)NEXP * TOKENS * 4;
  const size_t bf_off  = (hdr_end + 255) & ~(size_t)255;
  const size_t bf_need = bf_off + 3 * WELEMS * sizeof(__bf16);   // ~192.5 MB

  moe_zero_kernel<<<2048, 256, 0, stream>>>(out, TOKENS * HID, counts, sum_probs);
  moe_router_kernel<<<TOKENS / 256, 256, 0, stream>>>(xf, gate_w, combine, counts,
                                                      tok_list, sum_probs);
  moe_aux_kernel<<<1, 32, 0, stream>>>(counts, sum_probs, out + (size_t)TOKENS * HID);

  dim3 grid(TOKENS / BM, NEXP);
  if (ws_size >= bf_need) {
    // pre-convert weights to bf16 once per launch, then run bf16-weight FFN
    __bf16* wg_bf = (__bf16*)(ws + bf_off);
    __bf16* wu_bf = wg_bf + WELEMS;
    __bf16* wd_bf = wu_bf + WELEMS;
    const int n8 = (int)(WELEMS / 8);
    moe_cvt_bf16_kernel<<<4096, 256, 0, stream>>>(wg, wg_bf, n8);
    moe_cvt_bf16_kernel<<<4096, 256, 0, stream>>>(wu, wu_bf, n8);
    moe_cvt_bf16_kernel<<<4096, 256, 0, stream>>>(wd, wd_bf, n8);
    moe_ffn_kernel<__bf16><<<grid, NTHREADS, SMEM_BYTES, stream>>>(
        xf, wg_bf, wu_bf, wd_bf, counts, tok_list, combine, out);
  } else {
    // fallback: convert weights on the fly inside the FFN kernel
    moe_ffn_kernel<float><<<grid, NTHREADS, SMEM_BYTES, stream>>>(
        xf, wg, wu, wd, counts, tok_list, combine, out);
  }
}
